// LigerLMHeadGRPO_59090160058698
// MI455X (gfx1250) — matrix operand order repaired and checked by reading, not compiled, not run on hardware
//
#include <hip/hip_runtime.h>
#include <hip/hip_bf16.h>
#include <math.h>

// ---------------- problem constants ----------------
#define Bx 4
#define Tx 1024
#define Hx 2048
#define Vx 32000
#define NTOK (Bx * Tx)                // 4096
#define M_TILE 32                     // tokens per workgroup
#define MSUB 2                        // 16-row WMMA sub-tiles per wave
#define NSUB 2                        // 16-col WMMA sub-tiles per wave
#define NWAVES 8                      // 256 threads
#define NCOLS_PER_ITER (NWAVES * 16 * NSUB)  // 256
#define NVT (Vx / NCOLS_PER_ITER)     // 125 vocab tiles
#define NKK (Hx / 32)                 // 64 k-steps
#define NTILES (NTOK / M_TILE)        // 128

typedef __bf16 bf16x16 __attribute__((ext_vector_type(16)));
typedef __bf16 bf16x4  __attribute__((ext_vector_type(4)));
typedef float  v8f     __attribute__((ext_vector_type(8)));
typedef float  f32x4   __attribute__((ext_vector_type(4)));
typedef unsigned short u16x8 __attribute__((ext_vector_type(8)));

union AFrag { bf16x16 v; u16x8 h[2]; };
union BFrag { bf16x16 v; u16x8 h[2]; bf16x4 q[4]; };
static_assert(sizeof(AFrag) == 32, "frag size");
static_assert(sizeof(BFrag) == 32, "frag size");

// -------------------------------------------------------------------------
// Kernel 0: one-shot fp32 -> bf16 weight conversion (amortized once; bf16
// weights (2x128MB) then live mostly in the 192MB L2 during the GEMM)
// -------------------------------------------------------------------------
__global__ void cvt_weights_kernel(const float* __restrict__ src,
                                   __bf16* __restrict__ dst, int n4) {
    int i = blockIdx.x * blockDim.x + threadIdx.x;
    const int stride = gridDim.x * blockDim.x;
    for (; i < n4; i += stride) {
        f32x4 v = ((const f32x4*)src)[i];
        ((bf16x4*)dst)[i] = __builtin_convertvector(v, bf16x4);
    }
}

// -------------------------------------------------------------------------
// Kernel S: selected-token logit = dot(x[t], W[id[t]]) with bf16-rounded
// inputs (consistent with WMMA logits). One wave per (path, token).
// -------------------------------------------------------------------------
__global__ void __launch_bounds__(256)
grpo_sel_kernel(const float* __restrict__ x,
                const float* __restrict__ ref_input,
                const float* __restrict__ w,
                const float* __restrict__ ref_w,
                const int*   __restrict__ ids,
                float* __restrict__ selv)   // [2][NTOK]
{
    const int gw   = (blockIdx.x * blockDim.x + threadIdx.x) >> 5; // 0..2*NTOK-1
    const int lane = threadIdx.x & 31;
    const int path = gw >> 12;          // / NTOK
    const int t    = gw & (NTOK - 1);
    const float* __restrict__ X = path ? ref_input : x;
    const float* __restrict__ W = path ? ref_w : w;
    const float* xr = X + (size_t)t * Hx;
    const float* wr = W + (size_t)ids[t] * Hx;

    float acc = 0.0f;
    for (int j = lane * 4; j < Hx; j += 128) {
        f32x4 a  = *(const f32x4*)(xr + j);
        f32x4 b4 = *(const f32x4*)(wr + j);
        f32x4 af = __builtin_convertvector(__builtin_convertvector(a,  bf16x4), f32x4);
        f32x4 bf = __builtin_convertvector(__builtin_convertvector(b4, bf16x4), f32x4);
        acc += af.x * bf.x + af.y * bf.y + af.z * bf.z + af.w * bf.w;
    }
    #pragma unroll
    for (int m = 1; m < 32; m <<= 1) acc += __shfl_xor(acc, m, 32);
    if (lane == 0) selv[gw] = acc;
}

// -------------------------------------------------------------------------
// Kernel 1: per-32-token tile, logits via bf16 WMMA over the full vocab.
// Each wave owns a 32x32 (tokens x cols) macro-tile per iteration (MSUB=2,
// NSUB=2: A frags reused across 2 WMMAs -> 1 ds-load & 1 vmem-load per WMMA).
// Per-lane private online logsumexp, merged across lanes once at the end.
// PRE=true: B from pre-converted bf16 weights; PRE=false: cvt on the fly.
// -------------------------------------------------------------------------
template <bool PRE>
__global__ void __launch_bounds__(256)
grpo_logits_lse_kernel(const float* __restrict__ x,
                       const float* __restrict__ ref_input,
                       const float* __restrict__ w,
                       const float* __restrict__ ref_w,
                       const __bf16* __restrict__ wb,
                       const __bf16* __restrict__ ref_wb,
                       float* __restrict__ partM,  // [2][NTOK][NWAVES]
                       float* __restrict__ partS)  // [2][NTOK][NWAVES]
{
    __shared__ __align__(16) __bf16 sA[M_TILE * Hx];   // 128 KB bf16

    const int tile = blockIdx.x;
    const int path = blockIdx.y;
    const float*  __restrict__ X  = (path == 0) ? x  : ref_input;
    const float*  __restrict__ W  = (path == 0) ? w  : ref_w;
    const __bf16* __restrict__ WB = (path == 0) ? wb : ref_wb;

    const int tid  = threadIdx.x;
    const int wave = tid >> 5;
    const int lane = tid & 31;
    const int m0   = tile * M_TILE;

    // ---- stage 32 x 2048 activation tile into LDS as bf16 (hw packed cvt) ----
    for (int i = tid * 4; i < M_TILE * Hx; i += 256 * 4) {
        const int row = i >> 11;         // /Hx
        const int col = i & (Hx - 1);
        f32x4 vx = *(const f32x4*)&X[(size_t)(m0 + row) * Hx + col];
        *(bf16x4*)&sA[i] = __builtin_convertvector(vx, bf16x4);
    }
    __syncthreads();

    const int mlo   = lane & 15;   // row (A) / column (B) within a 16-tile
    const int half  = lane >> 4;
    const int aoff  = half * 16;   // A lane K base: 0..15 vs 16..31
    const int bkoff = half * 16;   // B lane K base: 0..15 vs 16..31

    // per-LANE private online logsumexp state (disjoint columns per lane)
    float m_run[MSUB][8], s_run[MSUB][8];
    #pragma unroll
    for (int s = 0; s < MSUB; ++s)
        #pragma unroll
        for (int r = 0; r < 8; ++r) { m_run[s][r] = -3.0e38f; s_run[s][r] = 0.0f; }

    for (int vt = 0; vt < NVT; ++vt) {
        const int n0 = vt * NCOLS_PER_ITER + wave * (16 * NSUB);
        size_t wrow[NSUB];
        #pragma unroll
        for (int n = 0; n < NSUB; ++n)
            wrow[n] = (size_t)(n0 + n * 16 + mlo) * Hx;

        v8f acc[NSUB][MSUB];
        #pragma unroll
        for (int n = 0; n < NSUB; ++n)
            #pragma unroll
            for (int s = 0; s < MSUB; ++s)
                acc[n][s] = (v8f){0.f, 0.f, 0.f, 0.f, 0.f, 0.f, 0.f, 0.f};

        #pragma unroll 2
        for (int kk = 0; kk < NKK; ++kk) {
            const int k0 = kk * 32;

            // A fragments: two 16-row sub-tiles from LDS (reused over NSUB)
            AFrag a[MSUB];
            #pragma unroll
            for (int s = 0; s < MSUB; ++s) {
                const u16x8* pa =
                    (const u16x8*)&sA[(s * 16 + mlo) * Hx + k0 + aoff];
                a[s].h[0] = pa[0];   // halfs k0+aoff .. +7
                a[s].h[1] = pa[1];   // halfs k0+aoff+8 .. +15
            }

            // B fragments: 16 contiguous K-elems from each of 2 weight rows
            BFrag b[NSUB];
            #pragma unroll
            for (int n = 0; n < NSUB; ++n) {
                if constexpr (PRE) {
                    const u16x8* wv = (const u16x8*)(WB + wrow[n] + k0 + bkoff);
                    b[n].h[0] = wv[0];
                    b[n].h[1] = wv[1];
                } else {
                    const f32x4* wv = (const f32x4*)(W + wrow[n] + k0 + bkoff);
                    b[n].q[0] = __builtin_convertvector(wv[0], bf16x4);
                    b[n].q[1] = __builtin_convertvector(wv[1], bf16x4);
                    b[n].q[2] = __builtin_convertvector(wv[2], bf16x4);
                    b[n].q[3] = __builtin_convertvector(wv[3], bf16x4);
                }
            }

            #pragma unroll
            for (int n = 0; n < NSUB; ++n)
                #pragma unroll
                for (int s = 0; s < MSUB; ++s)
                    acc[n][s] = __builtin_amdgcn_wmma_f32_16x16x32_bf16(
                        false, a[s].v, false, b[n].v, (short)0, acc[n][s],
                        false, false);
        }

        // ---- per-lane online logsumexp update (branchless, no shuffles) ----
        #pragma unroll
        for (int n = 0; n < NSUB; ++n) {
            #pragma unroll
            for (int s = 0; s < MSUB; ++s) {
                #pragma unroll
                for (int r = 0; r < 8; ++r) {
                    const float v  = acc[n][s][r];
                    const float nm = fmaxf(m_run[s][r], v);
                    s_run[s][r] = s_run[s][r] * __expf(m_run[s][r] - nm)
                                + __expf(v - nm);
                    m_run[s][r] = nm;
                }
            }
        }
    }

    // ---- one-time cross-lane merge of (max, sumexp) within 16-lane halves ----
    #pragma unroll
    for (int s = 0; s < MSUB; ++s) {
        #pragma unroll
        for (int r = 0; r < 8; ++r) {
            float m = m_run[s][r];
            float ss = s_run[s][r];
            #pragma unroll
            for (int msk = 1; msk <= 8; msk <<= 1) {
                const float om = __shfl_xor(m, msk, 32);
                const float os = __shfl_xor(ss, msk, 32);
                const float M  = fmaxf(m, om);
                ss = ss * __expf(m - M) + os * __expf(om - M);
                m  = M;
            }
            m_run[s][r] = m;
            s_run[s][r] = ss;
        }
    }

    // ---- spill per-wave logsumexp partials ----
    if (mlo == 0) {
        const size_t pbase = (size_t)path * NTOK * NWAVES;
        #pragma unroll
        for (int s = 0; s < MSUB; ++s) {
            #pragma unroll
            for (int r = 0; r < 8; ++r) {
                const int gtok = m0 + s * 16 + r + 8 * half;
                partM[pbase + (size_t)gtok * NWAVES + wave] = m_run[s][r];
                partS[pbase + (size_t)gtok * NWAVES + wave] = s_run[s][r];
            }
        }
    }
}

// -------------------------------------------------------------------------
// Kernel 2: combine 8 per-wave (max, sumexp) partials -> per-token logp
// -------------------------------------------------------------------------
__global__ void grpo_combine_kernel(const float* __restrict__ partM,
                                    const float* __restrict__ partS,
                                    const float* __restrict__ selv,
                                    float* __restrict__ logps)  // [2][NTOK]
{
    const int idx = blockIdx.x * blockDim.x + threadIdx.x;  // 0..2*NTOK-1
    if (idx >= 2 * NTOK) return;
    const float* pm = partM + (size_t)idx * NWAVES;
    const float* ps = partS + (size_t)idx * NWAVES;
    float M = pm[0];
    #pragma unroll
    for (int wv = 1; wv < NWAVES; ++wv) M = fmaxf(M, pm[wv]);
    float S = 0.0f;
    #pragma unroll
    for (int wv = 0; wv < NWAVES; ++wv) S += ps[wv] * __expf(pm[wv] - M);
    logps[idx] = selv[idx] - (M + __logf(S));
}

// -------------------------------------------------------------------------
// Kernel 3: GRPO clipped loss + KL, masked mean -> scalar
// -------------------------------------------------------------------------
__global__ void __launch_bounds__(256)
grpo_loss_kernel(const float* __restrict__ logps,      // [2][NTOK]
                 const int*   __restrict__ attn_mask,  // [NTOK]
                 const float* __restrict__ adv,        // [B]
                 const float* __restrict__ old_lp,     // [NTOK]
                 float* __restrict__ out)
{
    __shared__ float sn[256];
    __shared__ float sd[256];
    float num = 0.0f, den = 0.0f;
    for (int t = threadIdx.x; t < NTOK; t += 256) {
        const float lp  = logps[t];
        const float rlp = logps[NTOK + t];
        const float a   = adv[t >> 10];            // t / T
        const float mk  = (float)attn_mask[t];
        const float c1  = __expf(lp - old_lp[t]);
        const float c2  = fminf(fmaxf(c1, 0.8f), 1.2f);  // 1-EPS_LO, 1+EPS_HI
        float ptl = -fminf(c1 * a, c2 * a);
        const float d = rlp - lp;
        ptl += 0.1f * (__expf(d) - d - 1.0f);      // BETA * kl
        num += ptl * mk;
        den += mk;
    }
    sn[threadIdx.x] = num;
    sd[threadIdx.x] = den;
    __syncthreads();
    for (int s = 128; s > 0; s >>= 1) {
        if (threadIdx.x < s) {
            sn[threadIdx.x] += sn[threadIdx.x + s];
            sd[threadIdx.x] += sd[threadIdx.x + s];
        }
        __syncthreads();
    }
    if (threadIdx.x == 0) out[0] = sn[0] / fmaxf(sd[0], 1.0f);
}

// -------------------------------------------------------------------------
extern "C" void kernel_launch(void* const* d_in, const int* in_sizes, int n_in,
                              void* d_out, int out_size, void* d_ws, size_t ws_size,
                              hipStream_t stream) {
    (void)in_sizes; (void)n_in; (void)out_size;

    const float* x      = (const float*)d_in[0];
    const float* refin  = (const float*)d_in[1];
    const float* w      = (const float*)d_in[2];
    const float* refw   = (const float*)d_in[3];
    const int*   ids    = (const int*)d_in[4];
    const int*   mask   = (const int*)d_in[5];
    const float* adv    = (const float*)d_in[6];
    const float* old_lp = (const float*)d_in[7];
    float* out = (float*)d_out;

    // workspace layout
    float* ws     = (float*)d_ws;
    float* logps  = ws;                         // 2*NTOK
    float* selv   = ws + 2 * NTOK;              // 2*NTOK
    float* partM  = ws + 4 * NTOK;              // 2*NTOK*NWAVES
    float* partS  = ws + 4 * NTOK + 2 * NTOK * NWAVES;
    const size_t small_bytes =
        (size_t)(4 * NTOK + 4 * NTOK * NWAVES) * sizeof(float);   // 576 KB
    const size_t wb_elems = (size_t)Vx * Hx;                      // 65.5M
    const size_t wb_bytes = 2 * wb_elems * sizeof(__bf16);        // 262 MB

    const bool pre = (ws_size >= small_bytes + wb_bytes);

    dim3 grid1(NTILES, 2);
    if (pre) {
        __bf16* wb  = (__bf16*)((char*)d_ws + small_bytes);
        __bf16* rwb = wb + wb_elems;
        const int n4 = (int)(wb_elems / 4);
        cvt_weights_kernel<<<4096, 256, 0, stream>>>(w, wb, n4);
        cvt_weights_kernel<<<4096, 256, 0, stream>>>(refw, rwb, n4);
        grpo_logits_lse_kernel<true><<<grid1, 256, 0, stream>>>(
            x, refin, w, refw, wb, rwb, partM, partS);
    } else {
        grpo_logits_lse_kernel<false><<<grid1, 256, 0, stream>>>(
            x, refin, w, refw, nullptr, nullptr, partM, partS);
    }

    // selected-token logits (independent of the GEMM, tiny)
    grpo_sel_kernel<<<(2 * NTOK * 32) / 256, 256, 0, stream>>>(
        x, refin, w, refw, ids, selv);

    grpo_combine_kernel<<<(2 * NTOK + 255) / 256, 256, 0, stream>>>(
        partM, partS, selv, logps);

    grpo_loss_kernel<<<1, 256, 0, stream>>>(logps, mask, adv, old_lp, out);
}
